// ContextualAttention_43422119362842
// MI455X (gfx1250) — matrix operand order repaired and checked by reading, not compiled, not run on hardware
//
#include <hip/hip_runtime.h>
#include <hip/hip_bf16.h>

// ---------------------------------------------------------------------------
// Contextual attention, MI455X (gfx1250, wave32, WMMA bf16 K=32).
//   down(192->96), unfold 3x3 -> fgp/bgp [9216 x 864]
//   sim = bgp @ fgp^T (147 GFLOP), global-F normalize, softmax over keys,
//   wp = attn^T @ bgp (147 GFLOP), fold, mask, up(96->192).
// Flash-style fusion: sim (340MB fp32) is never materialized. Pass 1 streams
// sim tiles only for sum(sim^2); pass 2 fuses S-tile -> exp -> colsum ->
// E^T @ bgp per 16-query block. bgp (16MB bf16) lives in L2 (192MB).
// ---------------------------------------------------------------------------

#define H0 192
#define W0 192
#define HD 96
#define WD 96
#define HW (HD * WD)            // 9216
#define CCH 96
#define KDIM (CCH * 9)          // 864 = 27 * 32
#define KC 27                   // K chunks of 32
#define NB (HW / 16)            // 576 16-row blocks
#define NFT (KDIM / 16)         // 54 feature tiles
#define LAMBDA 10.0f
#define EPSN 1e-12f

typedef __attribute__((ext_vector_type(16))) __bf16 v16bf;
typedef __attribute__((ext_vector_type(8)))  __bf16 v8bf;
typedef __attribute__((ext_vector_type(8)))  float  v8f;

static __device__ inline unsigned short f2bf(float f) {
    union { float f; unsigned int u; } v; v.f = f;
    unsigned int u = v.u;
    u += 0x7fffu + ((u >> 16) & 1u);     // round-to-nearest-even
    return (unsigned short)(u >> 16);
}

static __device__ inline v16bf mk16(v8bf lo, v8bf hi) {
    return __builtin_shufflevector(lo, hi, 0,1,2,3,4,5,6,7,8,9,10,11,12,13,14,15);
}

static __device__ inline v8f wmma_bf16(v16bf a, v16bf b, v8f c) {
    return __builtin_amdgcn_wmma_f32_16x16x32_bf16(false, a, false, b,
                                                   (short)0, c, false, false);
}

// A-fragment (this file's contiguous k order; B uses the same): lane L -> row m = L&15, half = L>>4,
// element j -> k = half*16 + j  => one contiguous 16-elem run.
static __device__ inline v16bf load_afrag(const unsigned short* row, int kc, int half) {
    const unsigned short* p = row + kc * 32 + half * 16;
    return mk16(*(const v8bf*)p, *(const v8bf*)(p + 8));
}

// ------------------------------- kernels -----------------------------------

__global__ void init_kernel(float* sumsq) { *sumsq = 0.0f; }

// Downsample + mask + unfold directly from the 192x192 inputs into bf16
// patch matrices: fgp[p][k], bgp[p][k], bgpT[k][p].
__global__ void prep_kernel(const float* __restrict__ fg,
                            const float* __restrict__ bg,
                            const float* __restrict__ mask,
                            unsigned short* __restrict__ fgp,
                            unsigned short* __restrict__ bgp,
                            unsigned short* __restrict__ bgpT) {
    int tid = blockIdx.x * 256 + threadIdx.x;
    if (tid >= HW * KDIM) return;
    int k = tid % KDIM, p = tid / KDIM;
    int c = k / 9, r = k % 9, dy = r / 3, dx = r % 3;
    int y = p / WD, x = p % WD;
    int yy = y + dy - 1, xx = x + dx - 1;
    float fv = 0.f, bv = 0.f;
    if (yy >= 0 && yy < HD && xx >= 0 && xx < WD) {
        int src = (c * H0 + 2 * yy) * W0 + 2 * xx;
        float m = mask[(2 * yy) * W0 + 2 * xx];
        fv = fg[src] * m;
        bv = bg[src];
    }
    fgp[p * KDIM + k]  = f2bf(fv);
    bgp[p * KDIM + k]  = f2bf(bv);
    bgpT[k * HW + p]   = f2bf(bv);
}

// Pass 1: stream sim tiles (32x32 per wave) only for sum(sim^2).
__global__ __launch_bounds__(32)
void simsq_kernel(const unsigned short* __restrict__ fgp,
                  const unsigned short* __restrict__ bgp,
                  float* __restrict__ sumsq) {
    int lane = threadIdx.x, m = lane & 15, half = lane >> 4;
    int ib2 = blockIdx.x * 32;   // key rows
    int pb2 = blockIdx.y * 32;   // query cols
    v8f c00 = {}, c01 = {}, c10 = {}, c11 = {};
    const unsigned short* arow0 = bgp + (ib2 + m) * KDIM;
    const unsigned short* arow1 = arow0 + 16 * KDIM;
    const unsigned short* brow0 = fgp + (pb2 + m) * KDIM;
    const unsigned short* brow1 = brow0 + 16 * KDIM;
    for (int kc = 0; kc < KC; ++kc) {
        v16bf a0 = load_afrag(arow0, kc, half);
        v16bf a1 = load_afrag(arow1, kc, half);
        // B (K-major): lane -> col, k = half*16 + j => one 32B run per lane.
        v16bf b0 = *(const v16bf*)(brow0 + kc * 32 + half * 16);
        v16bf b1 = *(const v16bf*)(brow1 + kc * 32 + half * 16);
        c00 = wmma_bf16(a0, b0, c00);
        c01 = wmma_bf16(a0, b1, c01);
        c10 = wmma_bf16(a1, b0, c10);
        c11 = wmma_bf16(a1, b1, c11);
    }
    float v = 0.f;
#pragma unroll
    for (int r = 0; r < 8; ++r)
        v += c00[r]*c00[r] + c01[r]*c01[r] + c10[r]*c10[r] + c11[r]*c11[r];
    for (int off = 16; off > 0; off >>= 1) v += __shfl_down(v, off, 32);
    if (lane == 0) atomicAdd(sumsq, v);
}

__global__ void finalize_kernel(const float* sumsq, float* scale) {
    *scale = LAMBDA / fmaxf(sqrtf(*sumsq), EPSN);
}

// Pass 2 (fused): per 16-query block, loop over 576 key blocks:
//   S = bgp_kb @ fgp_qb^T  (split-K over 4 waves, LDS f32 reduce)
//   E = exp(scale * S);  colsum += sum_i E;  acc += E^T @ bgp_kb  (WMMA)
// Finally wp[q, :] = acc / colsum[q].
__global__ __launch_bounds__(128)
void attn_kernel(const unsigned short* __restrict__ fgp,
                 const unsigned short* __restrict__ bgp,
                 const unsigned short* __restrict__ bgpT,
                 const float* __restrict__ scale_p,
                 float* __restrict__ wp) {
    __shared__ float Ef32[256];                     // S tile [q][i]
    __shared__ __align__(32) unsigned short Ebf[16 * 32]; // exp tile, K-padded
    __shared__ float colsum[16];

    const int t = threadIdx.x;
    const int w = t >> 5, lane = t & 31, m = lane & 15, half = lane >> 4;
    const int pb = blockIdx.x;
    const float scale = *scale_p;

    for (int e = t; e < 512; e += 128) Ebf[e] = 0;   // upper K half stays 0
    if (t < 16) colsum[t] = 0.f;

    v8f acc[14];
#pragma unroll
    for (int i = 0; i < 14; ++i) acc[i] = v8f{};

    const unsigned short* qrow = fgp + (pb * 16 + m) * KDIM;
    __syncthreads();

    for (int ib = 0; ib < NB; ++ib) {
        const int ibase = ib * 16;
        // --- GEMM1 partial: this wave covers kc = w, w+4, ... (split-K) ---
        v8f s = {};
        const unsigned short* arow = bgp + (ibase + m) * KDIM;
#pragma unroll
        for (int q = 0; q < 7; ++q) {
            int kc = w + q * 4;
            if (kc < KC) {                       // wave-uniform branch
                v16bf a = load_afrag(arow, kc, half);
                v16bf b = *(const v16bf*)(qrow + kc * 32 + half * 16);
                s = wmma_bf16(a, b, s);
            }
        }
        Ef32[t] = 0.f; Ef32[t + 128] = 0.f;
        __syncthreads();
#pragma unroll
        for (int r = 0; r < 8; ++r)
            atomicAdd(&Ef32[m * 16 + r + half * 8], s[r]);   // [q=m][i]
        __syncthreads();
        // --- exp + column sums (|scale*S| <= LAMBDA, no max needed) ---
        for (int e = t; e < 256; e += 128) {
            int q = e >> 4, i = e & 15;
            float ev = __expf(scale * Ef32[e]);
            Ebf[q * 32 + i] = f2bf(ev);
            atomicAdd(&colsum[q], ev);
        }
        __syncthreads();
        // --- GEMM2: acc[q, feat] += E^T @ bgp_kb ---
        const v8bf* e0 = (const v8bf*)&Ebf[m * 32 + half * 16];
        const v8bf* e1 = (const v8bf*)&Ebf[m * 32 + half * 16 + 8];
        v16bf ea = mk16(*e0, *e1);               // A: M=q, K=i (i>=16 -> 0)
#pragma unroll
        for (int idx = 0; idx < 14; ++idx) {
            int ft = w + idx * 4;
            if (ft < NFT) {                      // wave-uniform branch
                v16bf b = {};
                if (half == 0)                   // K 16..31 padding is zero
                    b = *(const v16bf*)(bgpT + (ft * 16 + m) * HW + ibase);
                acc[idx] = wmma_bf16(ea, b, acc[idx]);
            }
        }
    }
    __syncthreads();
#pragma unroll
    for (int idx = 0; idx < 14; ++idx) {
        int ft = w + idx * 4;
        if (ft < NFT) {
#pragma unroll
            for (int r = 0; r < 8; ++r) {
                int pl = r + half * 8;           // C/D: M = r + 8*half, N = m
                wp[(pb * 16 + pl) * KDIM + ft * 16 + m] = acc[idx][r] / colsum[pl];
            }
        }
    }
}

// Fold (conv_transpose accumulate), mask, nearest 2x upsample.
__global__ void fold_kernel(const float* __restrict__ wp,
                            const float* __restrict__ mask,
                            float* __restrict__ out) {
    int tid = blockIdx.x * 256 + threadIdx.x;
    if (tid >= CCH * HD * WD) return;
    int xo = tid % WD, yo = (tid / WD) % HD, c = tid / (HD * WD);
    float s = 0.f;
#pragma unroll
    for (int dy = 0; dy < 3; ++dy)
#pragma unroll
        for (int dx = 0; dx < 3; ++dx) {
            int y = yo - dy + 1, x = xo - dx + 1;
            if (y >= 0 && y < HD && x >= 0 && x < WD)
                s += wp[(y * WD + x) * KDIM + c * 9 + dy * 3 + dx];
        }
    s *= mask[(2 * yo) * W0 + 2 * xo];
    int ob = (c * H0 + 2 * yo) * W0 + 2 * xo;
    out[ob] = s; out[ob + 1] = s; out[ob + W0] = s; out[ob + W0 + 1] = s;
}

// ------------------------------- launcher ----------------------------------

extern "C" void kernel_launch(void* const* d_in, const int* in_sizes, int n_in,
                              void* d_out, int out_size, void* d_ws, size_t ws_size,
                              hipStream_t stream) {
    const float* fg   = (const float*)d_in[0];
    const float* bg   = (const float*)d_in[1];
    const float* mask = (const float*)d_in[2];
    float* out = (float*)d_out;

    char* ws = (char*)d_ws;
    const size_t BF   = (size_t)HW * KDIM * 2;     // 15,925,248 B
    unsigned short* fgp  = (unsigned short*)(ws);
    unsigned short* bgp  = (unsigned short*)(ws + BF);
    unsigned short* bgpT = (unsigned short*)(ws + 2 * BF);
    float* wp    = (float*)(ws + 3 * BF);          // HW*KDIM f32
    float* sumsq = (float*)(ws + 3 * BF + (size_t)HW * KDIM * 4);
    float* scale = sumsq + 1;

    init_kernel<<<1, 1, 0, stream>>>(sumsq);

    int nprep = HW * KDIM;
    prep_kernel<<<(nprep + 255) / 256, 256, 0, stream>>>(fg, bg, mask, fgp, bgp, bgpT);

    simsq_kernel<<<dim3(NB / 2, NB / 2), 32, 0, stream>>>(fgp, bgp, sumsq);

    finalize_kernel<<<1, 1, 0, stream>>>(sumsq, scale);

    attn_kernel<<<NB, 128, 0, stream>>>(fgp, bgp, bgpT, scale, wp);

    int nfold = CCH * HD * WD;
    fold_kernel<<<(nfold + 255) / 256, 256, 0, stream>>>(wp, mask, out);
}
